// InformationGainRoutingBlock_36472862277803
// MI455X (gfx1250) — matrix operand order repaired and checked by reading, not compiled, not run on hardware
//
#include <hip/hip_runtime.h>

typedef __attribute__((ext_vector_type(16))) _Float16 v16h;
typedef __attribute__((ext_vector_type(8)))  _Float16 v8h;
typedef __attribute__((ext_vector_type(4)))  _Float16 v4h;
typedef __attribute__((ext_vector_type(8)))  float    v8f;
typedef __attribute__((ext_vector_type(4)))  float    v4f;

#define BN_EPS 0.001f
// B=64 H=W=56 C=256 -> OC=64, OH=OW=28, K = 9*256 = 2304 = 72 chunks of 32

// ---------------- Kernel 1: BN (inference) + fp32 -> f16 pack, NHWC ----------------
// Input is streamed once here -> non-temporal load; the f16 act buffer is re-read by the
// conv (~2.25x, overlapping 3x3 windows) -> regular store so it stays in the 192MB L2.
__global__ __launch_bounds__(256) void k_bn_pack(const float* __restrict__ in,
    const float* __restrict__ gamma, const float* __restrict__ beta,
    const float* __restrict__ mmean, const float* __restrict__ mvar,
    _Float16* __restrict__ act)
{
    size_t tid  = (size_t)blockIdx.x * 256 + threadIdx.x;   // one thread per 4 channels
    size_t base = tid * 4;
    int c = (int)(base & 255u);
    v4f x  = __builtin_nontemporal_load((const v4f*)(in + base));
    v4f g  = *(const v4f*)(gamma + c);
    v4f bt = *(const v4f*)(beta + c);
    v4f mu = *(const v4f*)(mmean + c);
    v4f va = *(const v4f*)(mvar + c);
    v4h o;
#pragma unroll
    for (int i = 0; i < 4; ++i) {
        float s = g[i] * rsqrtf(va[i] + BN_EPS);
        o[i] = (_Float16)((x[i] - mu[i]) * s + bt[i]);
    }
    *(v4h*)(act + base) = o;
}

// ------------- Kernel 2: pack conv weights (3,3,256,64) into B-fragment order -------------
// fragment (nt, kc, lane) at ((nt*72+kc)*32+lane)*16 halves; element e:
//   k = kc*32 + (e>>3)*16 + (lane>>4)*8 + (e&7)   (K index, channel-fastest)
//   n = nt*16 + (lane&15)          (output channel)
__global__ __launch_bounds__(256) void k_wpack(const float* __restrict__ w,
                                               _Float16* __restrict__ wp)
{
    int tid = blockIdx.x * 256 + threadIdx.x;      // 4*72*32*16 = 147456
    if (tid >= 4 * 72 * 32 * 16) return;
    int e    = tid & 15;
    int lane = (tid >> 4) & 31;
    int rest = tid >> 9;
    int kc   = rest % 72;
    int nt   = rest / 72;
    int k = kc * 32 + (e >> 3) * 16 + (lane >> 4) * 8 + (e & 7);
    int n = nt * 16 + (lane & 15);
    wp[tid] = (_Float16)w[k * 64 + n];
}

// ------------- Kernel 3: stride-2 3x3 conv as WMMA GEMM + fused mean-pool atomics -------------
// One wave = 16 output pixels x 64 output channels. 392 blocks x 8 waves = 3136 tiles = 64*49.
__global__ __launch_bounds__(256) void k_conv_wmma(const _Float16* __restrict__ act,
    const _Float16* __restrict__ wp, float* __restrict__ acc /* [64][64] */)
{
    const int lane = threadIdx.x & 31;
    const int wave = threadIdx.x >> 5;
    const int tile = blockIdx.x * 8 + wave;        // 0..3135
    const int b    = tile / 49;
    const int mt   = tile % 49;
    const int mrow = mt * 16 + (lane & 15);        // pixel 0..783
    const int oh = mrow / 28, ow = mrow % 28;
    const int k0 = (lane >> 4) * 8;                // A-frag K sub-offset (0 or 8)

    v8f c0 = {}, c1 = {}, c2 = {}, c3 = {};
    const _Float16* wl = wp + (size_t)lane * 16;

    for (int ky = 0; ky < 3; ++ky) {
        for (int kx = 0; kx < 3; ++kx) {
            const int slab = ky * 3 + kx;
            const int ih = 2 * oh + ky;            // SAME pad: lo=0, hi=1
            const int iw = 2 * ow + kx;
            const bool inb = (ih < 56) && (iw < 56);
            const _Float16* ap =
                act + (((size_t)((b * 56 + ih) * 56 + iw)) * 256 + k0);
#pragma unroll
            for (int cc = 0; cc < 8; ++cc) {
                const int kc = slab * 8 + cc;
                v16h a = {};
                if (inb) {
                    v8h lo = *(const v8h*)(ap + cc * 32);        // K k0..k0+7
                    v8h hi = *(const v8h*)(ap + cc * 32 + 16);   // K k0+16..k0+23
                    a = __builtin_shufflevector(lo, hi,
                        0,1,2,3,4,5,6,7,8,9,10,11,12,13,14,15);
                }
                v16h b0 = *(const v16h*)(wl + (size_t)((0 * 72 + kc) * 32) * 16);
                v16h b1 = *(const v16h*)(wl + (size_t)((1 * 72 + kc) * 32) * 16);
                v16h b2 = *(const v16h*)(wl + (size_t)((2 * 72 + kc) * 32) * 16);
                v16h b3 = *(const v16h*)(wl + (size_t)((3 * 72 + kc) * 32) * 16);
                c0 = __builtin_amdgcn_wmma_f32_16x16x32_f16(false, a, false, b0, (short)0, c0, false, false);
                c1 = __builtin_amdgcn_wmma_f32_16x16x32_f16(false, a, false, b1, (short)0, c1, false, false);
                c2 = __builtin_amdgcn_wmma_f32_16x16x32_f16(false, a, false, b2, (short)0, c2, false, false);
                c3 = __builtin_amdgcn_wmma_f32_16x16x32_f16(false, a, false, b3, (short)0, c3, false, false);
            }
        }
    }

    // Each lane holds 8 rows (pixels) for channel n = lane&15 per n-tile; both lane
    // halves cover all 16 rows. Sum rows locally, then atomic into per-batch channel sums.
    float s0 = 0.f, s1 = 0.f, s2 = 0.f, s3 = 0.f;
#pragma unroll
    for (int r = 0; r < 8; ++r) { s0 += c0[r]; s1 += c1[r]; s2 += c2[r]; s3 += c3[r]; }
    const int nl = lane & 15;
    float* ab = acc + b * 64;
    atomicAdd(ab + 0 * 16 + nl, s0);
    atomicAdd(ab + 1 * 16 + nl, s1);
    atomicAdd(ab + 2 * 16 + nl, s2);
    atomicAdd(ab + 3 * 16 + nl, s3);
}

// ------------- Kernel 4: mean finalize + FC + argmax -> routing_x, routes -------------
__global__ void k_finalize(const float* __restrict__ acc, const float* __restrict__ convb,
    const float* __restrict__ fcw, const float* __restrict__ fcb,
    float* __restrict__ rout, int* __restrict__ routes)
{
    int b = threadIdx.x;   // 64 threads, one batch each
    float r[4] = { fcb[0], fcb[1], fcb[2], fcb[3] };
    for (int c = 0; c < 64; ++c) {
        float m = acc[b * 64 + c] * (1.0f / 784.0f) + convb[c];
#pragma unroll
        for (int j = 0; j < 4; ++j) r[j] += m * fcw[c * 4 + j];
    }
#pragma unroll
    for (int j = 0; j < 4; ++j) rout[b * 4 + j] = r[j];
    int best = 0; float bv = r[0];                 // first-max tie-break like jnp.argmax
    if (r[1] > bv) { bv = r[1]; best = 1; }
    if (r[2] > bv) { bv = r[2]; best = 2; }
    if (r[3] > bv) { bv = r[3]; best = 3; }
    routes[b] = best;
}

// ------------- Kernel 5: channel gather of ORIGINAL inputs (float4 copies) -------------
// Pure streaming copy, nothing re-used -> non-temporal on both sides so it doesn't
// evict the act buffer (this kernel runs after conv, but replayed graphs loop).
__global__ __launch_bounds__(256) void k_gather(const float* __restrict__ in,
    const int* __restrict__ routes, float* __restrict__ out)
{
    size_t tid   = (size_t)blockIdx.x * 256 + threadIdx.x;  // 64*56*56*16 float4 groups
    size_t pixel = tid >> 4;
    int q = (int)(tid & 15);
    int b = (int)(pixel / 3136u);                            // 56*56
    int route = routes[b];
    v4f v = __builtin_nontemporal_load(
        (const v4f*)(in + pixel * 256 + (size_t)route * 64 + (size_t)q * 4));
    __builtin_nontemporal_store(v, (v4f*)(out + tid * 4));
}

extern "C" void kernel_launch(void* const* d_in, const int* in_sizes, int n_in,
                              void* d_out, int out_size, void* d_ws, size_t ws_size,
                              hipStream_t stream)
{
    const float* inputs = (const float*)d_in[0];
    const float* gamma  = (const float*)d_in[1];
    const float* beta   = (const float*)d_in[2];
    const float* mmean  = (const float*)d_in[3];
    const float* mvar   = (const float*)d_in[4];
    const float* convw  = (const float*)d_in[5];
    const float* convb  = (const float*)d_in[6];
    const float* fcw    = (const float*)d_in[7];
    const float* fcb    = (const float*)d_in[8];
    float* out = (float*)d_out;

    const size_t ACT_ELEMS = (size_t)64 * 56 * 56 * 256;   // 51,380,224 f16
    const size_t WP_ELEMS  = (size_t)4 * 72 * 32 * 16;     // 147,456 f16
    char* ws = (char*)d_ws;
    _Float16* act = (_Float16*)ws;
    _Float16* wp  = (_Float16*)(ws + ACT_ELEMS * 2);
    float*    acc = (float*)   (ws + ACT_ELEMS * 2 + WP_ELEMS * 2);
    int*   routes = (int*)     (ws + ACT_ELEMS * 2 + WP_ELEMS * 2 + 64 * 64 * 4);

    hipMemsetAsync(acc, 0, 64 * 64 * sizeof(float), stream);
    k_bn_pack  <<<dim3(50176), dim3(256), 0, stream>>>(inputs, gamma, beta, mmean, mvar, act);
    k_wpack    <<<dim3((147456 + 255) / 256), dim3(256), 0, stream>>>(convw, wp);
    k_conv_wmma<<<dim3(392), dim3(256), 0, stream>>>(act, wp, acc);
    k_finalize <<<dim3(1), dim3(64), 0, stream>>>(acc, convb, fcw, fcb,
                                                  out + (size_t)64 * 56 * 56 * 64, routes);
    k_gather   <<<dim3(12544), dim3(256), 0, stream>>>(inputs, routes, out);
}
